// Full_GSA_SC_30545807409735
// MI455X (gfx1250) — matrix operand order repaired and checked by reading, not compiled, not run on hardware
//
#include <hip/hip_runtime.h>

typedef __bf16 bf16_t;
typedef __attribute__((ext_vector_type(16))) __bf16 v16bf;
typedef __attribute__((ext_vector_type(8)))  __bf16 v8bf;
typedef __attribute__((ext_vector_type(8)))  float  v8f;
typedef __attribute__((ext_vector_type(4)))  int    v4i;

constexpr int B_  = 4;
constexpr int H_  = 48;
constexpr int W_  = 48;
constexpr int C_  = 256;
constexpr int NH  = 8;
constexpr int D_  = 32;
constexpr int N_  = H_ * W_;        // 2304
constexpr int M_  = B_ * N_;        // 9216

#define WMMA_BF16(a, b, c) \
  __builtin_amdgcn_wmma_f32_16x16x32_bf16(false, (a), false, (b), (short)0, (c), false, false)

#if defined(__has_builtin)
#if __has_builtin(__builtin_amdgcn_global_load_async_to_lds_b128) && \
    __has_builtin(__builtin_amdgcn_s_wait_asynccnt)
#define HAVE_ASYNC_LDS 1
#endif
#endif

#ifndef HAVE_ASYNC_LDS
#define HAVE_ASYNC_LDS 0
#endif

__device__ __forceinline__ v8bf ld8g(const bf16_t* p) { return *(const v8bf*)p; }

__device__ __forceinline__ v16bf pack16(v8bf lo, v8bf hi) {
  v16bf r;
#pragma unroll
  for (int i = 0; i < 8; ++i) { r[i] = lo[i]; r[i + 8] = hi[i]; }
  return r;
}

// ---------------------------------------------------------------------------
// fp32 -> bf16 elementwise
// ---------------------------------------------------------------------------
__global__ __launch_bounds__(256) void cvt_bf16_kernel(const float* __restrict__ src,
                                                       bf16_t* __restrict__ dst, int n) {
  int i = blockIdx.x * blockDim.x + threadIdx.x;
  if (i < n) dst[i] = (bf16_t)src[i];
}

// transpose 256x256 fp32 -> bf16 : dst[c*256+k] = src[k*256+c]
__global__ __launch_bounds__(256) void cvt_wt_kernel(const float* __restrict__ src,
                                                     bf16_t* __restrict__ dst) {
  int i = blockIdx.x * blockDim.x + threadIdx.x;   // 65536
  int c = i >> 8, k = i & 255;
  dst[c * 256 + k] = (bf16_t)src[k * 256 + c];
}

// ---------------------------------------------------------------------------
// WMMA GEMM:  out[M x 256] = A_bf16[M x 256] @ Wt_bf16(NxK, transposed) ; (acc+bias)*scale
// block = 128 threads (4 waves), each wave computes a 16x64 tile.
// The block's 64x256 bf16 weight slab is staged into LDS once (async copy,
// ASYNCcnt-tracked on gfx1250), then all 4 waves read B tiles via ds_load_b128.
// LDS rows padded to 264 bf16 (132 dwords) -> 16-lane tile reads are bank-
// conflict free (bank advances by 4 per lane).
// grid = (M/64, 4)
// ---------------------------------------------------------------------------
constexpr int WROW = 264;   // padded LDS row length in bf16 elements

__global__ __launch_bounds__(128) void gemm256_kernel(const bf16_t* __restrict__ A,
                                                      const bf16_t* __restrict__ Wt,
                                                      const float* __restrict__ bias,
                                                      float scale,
                                                      float* __restrict__ out) {
  __shared__ __align__(16) bf16_t wlds[64 * WROW];

  int lane = threadIdx.x & 31;
  int wave = threadIdx.x >> 5;
  int g  = lane & 15;
  int hi = lane >> 4;
  int mtile = (blockIdx.x * 4 + wave) * 16;
  int cbase = blockIdx.y * 64;

  // ---- stage the 64x256 weight slab into LDS (2048 chunks of 16B)
#pragma unroll
  for (int it = 0; it < 16; ++it) {
    int ch  = threadIdx.x + it * 128;
    int row = ch >> 5;          // 0..63
    int sub = ch & 31;          // 0..31 (16B chunks within a 512B row)
    const bf16_t* src = Wt + (size_t)(cbase + row) * 256 + sub * 8;
    bf16_t*       dst = wlds + row * WROW + sub * 8;
#if HAVE_ASYNC_LDS
    __builtin_amdgcn_global_load_async_to_lds_b128(
        (__attribute__((address_space(1))) v4i*)src,
        (__attribute__((address_space(3))) v4i*)dst, 0, 0);
#else
    *(v8bf*)dst = *(const v8bf*)src;
#endif
  }
#if HAVE_ASYNC_LDS
  __builtin_amdgcn_s_wait_asynccnt(0);
#endif
  __syncthreads();

  v8f acc[4];
#pragma unroll
  for (int t = 0; t < 4; ++t) acc[t] = v8f{0.f, 0.f, 0.f, 0.f, 0.f, 0.f, 0.f, 0.f};

  const bf16_t* arow = A + (size_t)(mtile + g) * 256;

  for (int k0 = 0; k0 < 256; k0 += 32) {
    // A-matrix 16x32 bf16: low lanes hold K {0..7,16..23}, high lanes {8..15,24..31}
    v16bf a = pack16(ld8g(arow + k0 + hi * 8), ld8g(arow + k0 + 16 + hi * 8));
#pragma unroll
    for (int t = 0; t < 4; ++t) {
      // B-matrix 32x16 bf16: low lanes hold K {0..7,16..23}, high lanes {8..15,24..31}
      const bf16_t* wrow = wlds + (t * 16 + g) * WROW + k0 + hi * 8;
      v16bf bm = pack16(*(const v8bf*)wrow, *(const v8bf*)(wrow + 16));
      acc[t] = WMMA_BF16(a, bm, acc[t]);
    }
  }

#pragma unroll
  for (int t = 0; t < 4; ++t) {
    int ccol = cbase + t * 16 + g;
    float bv = bias[ccol];
#pragma unroll
    for (int r = 0; r < 8; ++r) {
      out[(size_t)(mtile + r + hi * 8) * 256 + ccol] = (acc[t][r] + bv) * scale;
    }
  }
}

// ---------------------------------------------------------------------------
// RoPE + repack: q,k -> bf16 (B,NH,N,32) rotated; v -> bf16 transposed (B,NH,32,N)
// ---------------------------------------------------------------------------
__global__ __launch_bounds__(256) void rope_pack_kernel(const float* __restrict__ qf,
                                                        const float* __restrict__ kf,
                                                        const float* __restrict__ vf,
                                                        const float* __restrict__ sinp,
                                                        const float* __restrict__ cosp,
                                                        bf16_t* __restrict__ qb,
                                                        bf16_t* __restrict__ kb,
                                                        bf16_t* __restrict__ vt) {
  int idx = blockIdx.x * blockDim.x + threadIdx.x;
  if (idx >= M_ * C_) return;
  int c = idx & 255;
  int n = (idx >> 8) % N_;
  int b = idx / (256 * N_);
  int h = c >> 5, j = c & 31;

  float sn = sinp[n * 32 + j];
  float cs = cosp[n * 32 + j];
  int pidx = idx + ((j & 1) ? -1 : 1);   // rotation partner within the head dim
  float sgn = (j & 1) ? sn : -sn;

  float qr = qf[idx] * cs + qf[pidx] * sgn;
  float kr = kf[idx] * cs + kf[pidx] * sgn;

  size_t o = ((size_t)(b * NH + h) * N_ + n) * 32 + j;
  qb[o] = (bf16_t)qr;
  kb[o] = (bf16_t)kr;
  vt[((size_t)(b * NH + h) * 32 + j) * N_ + n] = (bf16_t)vf[idx];
}

// ---------------------------------------------------------------------------
// Flash attention: one wave per (b,h,16 query rows).  grid=(N/16, B*NH), block=32
// ---------------------------------------------------------------------------
__global__ __launch_bounds__(32) void attn_kernel(const bf16_t* __restrict__ qb,
                                                  const bf16_t* __restrict__ kb,
                                                  const bf16_t* __restrict__ vt,
                                                  const float* __restrict__ mask,
                                                  float* __restrict__ outp) {
  __shared__ __align__(32) bf16_t plds[16 * 32];

  int lane = threadIdx.x;
  int g  = lane & 15;
  int hi = lane >> 4;
  int bh = blockIdx.y;
  int b = bh >> 3, h = bh & 7;
  int qbase = blockIdx.x * 16;

  const bf16_t* qp = qb + (size_t)(b * NH + h) * N_ * 32;
  const bf16_t* kp = kb + (size_t)(b * NH + h) * N_ * 32;
  const bf16_t* vp = vt + (size_t)(b * NH + h) * 32 * N_;
  const float*  mp = mask + (size_t)h * N_ * N_;

  // Load the Q tile once (A-matrix layout)
  const bf16_t* qr = qp + (size_t)(qbase + g) * 32;
  v16bf aq = pack16(ld8g(qr + hi * 8), ld8g(qr + 16 + hi * 8));

  float mi[8], li[8];
#pragma unroll
  for (int r = 0; r < 8; ++r) { mi[r] = -1e30f; li[r] = 0.f; }
  v8f o0 = v8f{0.f,0.f,0.f,0.f,0.f,0.f,0.f,0.f};
  v8f o1 = o0;
  const float LOG2E = 1.4426950408889634f;

  for (int k0 = 0; k0 < N_; k0 += 32) {
    // ---- S = Q @ K^T for 32 keys (two 16x16 tiles)
    v8f s0 = v8f{0.f,0.f,0.f,0.f,0.f,0.f,0.f,0.f};
    v8f s1 = s0;
    {
      const bf16_t* kr0 = kp + (size_t)(k0 + g) * 32 + hi * 8;
      v16bf bm = pack16(ld8g(kr0), ld8g(kr0 + 16));
      s0 = WMMA_BF16(aq, bm, s0);
    }
    {
      const bf16_t* kr1 = kp + (size_t)(k0 + 16 + g) * 32 + hi * 8;
      v16bf bm = pack16(ld8g(kr1), ld8g(kr1 + 16));
      s1 = WMMA_BF16(aq, bm, s1);
    }
    // ---- add mask
#pragma unroll
    for (int r = 0; r < 8; ++r) {
      size_t row = (size_t)(qbase + r + hi * 8) * N_;
      s0[r] += mp[row + k0 + g];
      s1[r] += mp[row + k0 + 16 + g];
    }
    if (k0 + 32 < N_) {
      __builtin_prefetch(mp + (size_t)(qbase + hi * 8) * N_ + k0 + 32, 0, 1);
    }
    // ---- online softmax: row max across the 16-lane group
    float nm[8], fac[8], rs[8];
#pragma unroll
    for (int r = 0; r < 8; ++r) {
      float mx = fmaxf(s0[r], s1[r]);
      mx = fmaxf(mx, __shfl_xor(mx, 1, 32));
      mx = fmaxf(mx, __shfl_xor(mx, 2, 32));
      mx = fmaxf(mx, __shfl_xor(mx, 4, 32));
      mx = fmaxf(mx, __shfl_xor(mx, 8, 32));
      nm[r]  = fmaxf(mi[r], mx);
      fac[r] = exp2f((mi[r] - nm[r]) * LOG2E);
    }
#pragma unroll
    for (int r = 0; r < 8; ++r) {
      float p0 = exp2f((s0[r] - nm[r]) * LOG2E);
      float p1 = exp2f((s1[r] - nm[r]) * LOG2E);
      plds[(r + hi * 8) * 32 + g]      = (bf16_t)p0;
      plds[(r + hi * 8) * 32 + 16 + g] = (bf16_t)p1;
      float ps = p0 + p1;
      ps += __shfl_xor(ps, 1, 32);
      ps += __shfl_xor(ps, 2, 32);
      ps += __shfl_xor(ps, 4, 32);
      ps += __shfl_xor(ps, 8, 32);
      rs[r] = ps;
    }
    __syncthreads();
    // re-layout P into A-matrix registers
    const bf16_t* pr = plds + g * 32;
    v16bf ap = pack16(*(const v8bf*)(pr + hi * 8), *(const v8bf*)(pr + 16 + hi * 8));
    __syncthreads();
    // ---- rescale running accumulators
#pragma unroll
    for (int r = 0; r < 8; ++r) {
      o0[r] *= fac[r];
      o1[r] *= fac[r];
      li[r] = li[r] * fac[r] + rs[r];
      mi[r] = nm[r];
    }
    // ---- O += P @ V  (two d-column tiles)
    {
      const bf16_t* vr = vp + (size_t)g * N_ + k0 + hi * 8;
      v16bf bm = pack16(ld8g(vr), ld8g(vr + 16));
      o0 = WMMA_BF16(ap, bm, o0);
    }
    {
      const bf16_t* vr = vp + (size_t)(16 + g) * N_ + k0 + hi * 8;
      v16bf bm = pack16(ld8g(vr), ld8g(vr + 16));
      o1 = WMMA_BF16(ap, bm, o1);
    }
  }

  // ---- finalize: divide by softmax denominator, write (B,N,C) fp32
#pragma unroll
  for (int r = 0; r < 8; ++r) {
    float inv = 1.0f / li[r];
    size_t row = (size_t)(b * N_ + qbase + r + hi * 8) * 256 + h * 32;
    outp[row + g]      = o0[r] * inv;
    outp[row + 16 + g] = o1[r] * inv;
  }
}

// ---------------------------------------------------------------------------
// Depthwise 5x5 conv (lepe) + add attention output, convert to bf16 A matrix
// ---------------------------------------------------------------------------
__global__ __launch_bounds__(256) void lepe_add_kernel(const float* __restrict__ vf,
                                                       const float* __restrict__ dww,
                                                       const float* __restrict__ dwb,
                                                       const float* __restrict__ attn,
                                                       bf16_t* __restrict__ afin) {
  int idx = blockIdx.x * blockDim.x + threadIdx.x;
  if (idx >= M_ * C_) return;
  int c  = idx & 255;
  int hw = idx >> 8;
  int w  = hw % W_;
  int hh = (hw / W_) % H_;
  int b  = idx / (256 * N_);

  float s = dwb[c];
#pragma unroll
  for (int ky = 0; ky < 5; ++ky) {
    int y = hh + ky - 2;
    if (y < 0 || y >= H_) continue;
#pragma unroll
    for (int kx = 0; kx < 5; ++kx) {
      int x = w + kx - 2;
      if (x < 0 || x >= W_) continue;
      s += vf[((size_t)(b * H_ + y) * W_ + x) * 256 + c] * dww[(ky * 5 + kx) * 256 + c];
    }
  }
  afin[idx] = (bf16_t)(attn[idx] + s);
}

// ---------------------------------------------------------------------------
extern "C" void kernel_launch(void* const* d_in, const int* in_sizes, int n_in,
                              void* d_out, int out_size, void* d_ws, size_t ws_size,
                              hipStream_t stream) {
  const float* x    = (const float*)d_in[0];
  const float* sinp = (const float*)d_in[1];
  const float* cosp = (const float*)d_in[2];
  const float* mask = (const float*)d_in[3];
  const float* Wq   = (const float*)d_in[4];
  const float* bq   = (const float*)d_in[5];
  const float* Wk   = (const float*)d_in[6];
  const float* bk   = (const float*)d_in[7];
  const float* Wv   = (const float*)d_in[8];
  const float* bv   = (const float*)d_in[9];
  const float* dww  = (const float*)d_in[10];
  const float* dwb  = (const float*)d_in[11];
  const float* Wo   = (const float*)d_in[12];
  const float* bo   = (const float*)d_in[13];
  float* out = (float*)d_out;

  char* ws = (char*)d_ws;
  size_t off = 0;
  auto alloc = [&](size_t bytes) -> void* {
    void* p = ws + off;
    off += (bytes + 255) & ~(size_t)255;
    return p;
  };

  float*  qf   = (float*)alloc((size_t)M_ * C_ * 4);
  float*  kf   = (float*)alloc((size_t)M_ * C_ * 4);
  float*  vf   = (float*)alloc((size_t)M_ * C_ * 4);
  float*  attn = (float*)alloc((size_t)M_ * C_ * 4);
  bf16_t* xb   = (bf16_t*)alloc((size_t)M_ * C_ * 2);
  bf16_t* wqt  = (bf16_t*)alloc((size_t)C_ * C_ * 2);
  bf16_t* wkt  = (bf16_t*)alloc((size_t)C_ * C_ * 2);
  bf16_t* wvt  = (bf16_t*)alloc((size_t)C_ * C_ * 2);
  bf16_t* wot  = (bf16_t*)alloc((size_t)C_ * C_ * 2);
  bf16_t* qb   = (bf16_t*)alloc((size_t)M_ * C_ * 2);
  bf16_t* kb   = (bf16_t*)alloc((size_t)M_ * C_ * 2);
  bf16_t* vtb  = (bf16_t*)alloc((size_t)M_ * C_ * 2);
  bf16_t* afin = (bf16_t*)alloc((size_t)M_ * C_ * 2);

  const int total = M_ * C_;
  const float kscale = 0.17677669529663687f;   // 32^-0.5

  // 1) conversions
  cvt_bf16_kernel<<<total / 256, 256, 0, stream>>>(x, xb, total);
  cvt_wt_kernel<<<(C_ * C_) / 256, 256, 0, stream>>>(Wq, wqt);
  cvt_wt_kernel<<<(C_ * C_) / 256, 256, 0, stream>>>(Wk, wkt);
  cvt_wt_kernel<<<(C_ * C_) / 256, 256, 0, stream>>>(Wv, wvt);
  cvt_wt_kernel<<<(C_ * C_) / 256, 256, 0, stream>>>(Wo, wot);

  // 2) QKV projections (WMMA GEMMs)
  dim3 ggrid(M_ / 64, 4);
  gemm256_kernel<<<ggrid, 128, 0, stream>>>(xb, wqt, bq, 1.0f, qf);
  gemm256_kernel<<<ggrid, 128, 0, stream>>>(xb, wkt, bk, kscale, kf);
  gemm256_kernel<<<ggrid, 128, 0, stream>>>(xb, wvt, bv, 1.0f, vf);

  // 3) RoPE + repack to bf16 attention layouts
  rope_pack_kernel<<<total / 256, 256, 0, stream>>>(qf, kf, vf, sinp, cosp, qb, kb, vtb);

  // 4) flash attention with mask
  dim3 agrid(N_ / 16, B_ * NH);
  attn_kernel<<<agrid, 32, 0, stream>>>(qb, kb, vtb, mask, attn);

  // 5) lepe depthwise conv + residual add -> bf16
  lepe_add_kernel<<<total / 256, 256, 0, stream>>>(vf, dww, dwb, attn, afin);

  // 6) output projection -> d_out
  gemm256_kernel<<<ggrid, 128, 0, stream>>>(afin, wot, bo, 1.0f, out);
}
